// DecoderLSTMPred_85572928406172
// MI455X (gfx1250) — matrix operand order repaired and checked by reading, not compiled, not run on hardware
//
#include <hip/hip_runtime.h>

typedef __attribute__((ext_vector_type(8)))  __bf16 v8bf;
typedef __attribute__((ext_vector_type(16))) __bf16 v16bf;
typedef __attribute__((ext_vector_type(8)))  float  v8f;
typedef int v4i __attribute__((vector_size(16)));    // matches builtin proto

// Problem dims (match reference)
constexpr int B_ = 128, S_ = 128, F_ = 512, H_ = 1024;
constexpr int KCAT = F_ + H_;        // 1536 (x | h concatenated K)
constexpr int NG   = 4 * H_;         // 4096 gate columns
constexpr int KT_G = KCAT / 32;      // 48 K-tiles for gates GEMM
constexpr int KT_F = H_ / 32;        // 32 K-tiles for fc GEMM

#define DEV __device__ __forceinline__

// ---- CDNA5 async global->LDS path (guarded; falls back to sync staging) ----
#if defined(__has_builtin)
#if __has_builtin(__builtin_amdgcn_global_load_async_to_lds_b128) && \
    __has_builtin(__builtin_amdgcn_global_load_async_to_lds_b32)
#define USE_ASYNC_LDS 1
#endif
#endif

#if defined(USE_ASYNC_LDS)
typedef __attribute__((address_space(1))) v4i GV4;
typedef __attribute__((address_space(3))) v4i LV4;
typedef __attribute__((address_space(1))) int GI;
typedef __attribute__((address_space(3))) int LI;

DEV void async_cp_b128(const void* g, void* l) {
  __builtin_amdgcn_global_load_async_to_lds_b128((GV4*)g, (LV4*)l, 0, 0);
}
DEV void async_cp_b32(const void* g, void* l) {
  __builtin_amdgcn_global_load_async_to_lds_b32((GI*)g, (LI*)l, 0, 0);
}
#if __has_builtin(__builtin_amdgcn_s_wait_asynccnt)
#define WAIT_ASYNC(n) __builtin_amdgcn_s_wait_asynccnt(n)
#else
#define WAIT_ASYNC(n) asm volatile("s_wait_asynccnt %0" ::"i"(n) : "memory")
#endif
#endif

DEV float sigmoidf_(float x) { return 1.0f / (1.0f + __expf(-x)); }

DEV unsigned short to_bf16_bits(float v) {
  __bf16 b = (__bf16)v;
  return *reinterpret_cast<unsigned short*>(&b);
}

// A operand (16x32 bf16): lane L holds row m=L&15; K halves by sel=L>>4:
// elements 0..15 = K [k0 + sel*16, +16) (K-contiguous per lane)
DEV v16bf load_a16(const __bf16* rowbase_k0, int sel) {
  const v8bf lo = *reinterpret_cast<const v8bf*>(rowbase_k0 + sel * 16);
  const v8bf hi = *reinterpret_cast<const v8bf*>(rowbase_k0 + sel * 16 + 8);
  v16bf a;
#pragma unroll
  for (int i = 0; i < 8; ++i) { a[i] = lo[i]; a[8 + i] = hi[i]; }
  return a;
}

// B operand: packed so lane L's 16 values are contiguous at p = tile + L*16
DEV v16bf load_b16(const __bf16* p) {
  const v8bf lo = *reinterpret_cast<const v8bf*>(p);
  const v8bf hi = *reinterpret_cast<const v8bf*>(p + 8);
  v16bf b;
#pragma unroll
  for (int i = 0; i < 8; ++i) { b[i] = lo[i]; b[8 + i] = hi[i]; }
  return b;
}

DEV v8f wmma_bf16(v16bf a, v16bf b, v8f c) {
  return __builtin_amdgcn_wmma_f32_16x16x32_bf16(false, a, false, b, (short)0, c,
                                                 false, false);
}

// ---------------- prep kernels ----------------

// Pack [W_ih | W_hh] -> bf16 B-operand tiles: tile index (nt*KT_G + kt),
// intra-tile order: lane(0..31) * 16 + j, with n = nt*16 + (lane&15),
// k = kt*32 + (lane>>4)*16 + j   (K-contiguous per lane).
__global__ __launch_bounds__(256) void k_pack_gates(
    const float* __restrict__ Wih, const float* __restrict__ Whh,
    unsigned short* __restrict__ Wg) {
  const int p = blockIdx.x * 256 + threadIdx.x;     // < 4096*1536
  const int tile = p >> 9, r = p & 511;
  const int lane = r >> 4, j = r & 15;
  const int kt = tile % KT_G, nt = tile / KT_G;
  const int n = nt * 16 + (lane & 15);
  const int k = kt * 32 + (lane >> 4) * 16 + j;
  const float v = (k < F_) ? Wih[(size_t)n * F_ + k]
                           : Whh[(size_t)n * H_ + (k - F_)];
  Wg[p] = to_bf16_bits(v);
}

__global__ __launch_bounds__(256) void k_pack_fc(const float* __restrict__ fcW,
                                                 unsigned short* __restrict__ Wf) {
  const int p = blockIdx.x * 256 + threadIdx.x;     // < 512*1024
  const int tile = p >> 9, r = p & 511;
  const int lane = r >> 4, j = r & 15;
  const int kt = tile % KT_F, nt = tile / KT_F;
  const int n = nt * 16 + (lane & 15);
  const int k = kt * 32 + (lane >> 4) * 16 + j;
  Wf[p] = to_bf16_bits(fcW[(size_t)n * H_ + k]);
}

__global__ __launch_bounds__(256) void k_init(
    const float* __restrict__ input, const float* __restrict__ h0,
    const float* __restrict__ c0, const float* __restrict__ bih,
    const float* __restrict__ bhh, float* __restrict__ bias,
    unsigned short* __restrict__ xb, unsigned short* __restrict__ hb0,
    float* __restrict__ cb) {
  const int i = blockIdx.x * 256 + threadIdx.x;     // < 128*1024
  if (i < NG) bias[i] = bih[i] + bhh[i];
  if (i < B_ * F_) {
    const int b = i / F_, f = i % F_;
    xb[i] = to_bf16_bits(input[(size_t)b * S_ * F_ + f]);   // input[:,0,:]
  }
  if (i < B_ * H_) {
    hb0[i] = to_bf16_bits(h0[i]);
    cb[i] = c0[i];
  }
}

// ---------------- per-step kernels ----------------

// gates = [x|h] @ Wcat + b ; LSTM cell ; write h_new (bf16), c in-place (f32)
// grid = 64 (hidden 16-col tiles), block = 256 (8 waves = 8 M-tiles)
__global__ __launch_bounds__(256) void k_gates_cell(
    const unsigned short* __restrict__ xbits,   // [128,512]  bf16
    const unsigned short* __restrict__ hbits,   // [128,1024] bf16 (prev step)
    const unsigned short* __restrict__ Wgbits,  // packed gate weights
    const float* __restrict__ bias,             // [4096]
    float* __restrict__ cbuf,                   // [128,1024] f32 (in-place)
    unsigned short* __restrict__ hnew) {        // [128,1024] bf16
  const int tid = threadIdx.x;
  const int wave = tid >> 5, lane = tid & 31;
  const int nth = blockIdx.x;                  // hidden n-tile 0..63
  const int sel = lane >> 4, lr = lane & 15;
  const int m = wave * 16 + lr;                // A row for this lane

  const __bf16* x = reinterpret_cast<const __bf16*>(xbits);
  const __bf16* h = reinterpret_cast<const __bf16*>(hbits);
  const __bf16* w = reinterpret_cast<const __bf16*>(Wgbits);

  // cooperative B-tile load mapping: 256 threads x 16B = 4KB (no divergence)
  const int tg = tid >> 6, toff = tid & 63;

  v8f acc0 = {}, acc1 = {}, acc2 = {}, acc3 = {};

#if defined(USE_ASYNC_LDS)
  // Triple-buffered async pipeline: issue tile kt+1 while computing kt.
  // s_wait_asynccnt(1) + ONE barrier per iteration; the slot being written
  // at iter kt was last read at iter kt-2 (one barrier in between).
  __shared__ __align__(16) unsigned short sB[3][4 * 512];   // 3 x 4KB
  const __bf16* sb = reinterpret_cast<const __bf16*>(&sB[0][0]);

  auto issue = [&](int kt, int slot) {
    const __bf16* src =
        w + ((size_t)((tg * 64 + nth) * KT_G + kt)) * 512 + toff * 8;
    async_cp_b128(src, &sB[slot][tid * 8]);
  };

  issue(0, 0);
  int cur = 0;
  for (int kt = 0; kt < KT_G; ++kt) {
    int nxt = cur + 1; if (nxt == 3) nxt = 0;
    if (kt + 1 < KT_G) {
      issue(kt + 1, nxt);
      WAIT_ASYNC(1);            // current tile's copy complete (per wave)
    } else {
      WAIT_ASYNC(0);
    }
    __syncthreads();            // all waves' portions visible

    const __bf16* arow = (kt < 16) ? (x + (size_t)m * F_ + kt * 32)
                                   : (h + (size_t)m * H_ + (kt - 16) * 32);
    const __bf16* base = sb + cur * 2048;
    const v16bf a = load_a16(arow, sel);
    const v16bf b0 = load_b16(base + 0 * 512 + lane * 16);
    const v16bf b1 = load_b16(base + 1 * 512 + lane * 16);
    const v16bf b2 = load_b16(base + 2 * 512 + lane * 16);
    const v16bf b3 = load_b16(base + 3 * 512 + lane * 16);
    acc0 = wmma_bf16(a, b0, acc0);
    acc1 = wmma_bf16(a, b1, acc1);
    acc2 = wmma_bf16(a, b2, acc2);
    acc3 = wmma_bf16(a, b3, acc3);
    cur = nxt;
  }
#else
  // Fallback: synchronous staging through VGPRs (known-good path)
  __shared__ __align__(16) unsigned short sB[4 * 512];
  const __bf16* sb = reinterpret_cast<const __bf16*>(sB);

  for (int kt = 0; kt < KT_G; ++kt) {
    const __bf16* src =
        w + ((size_t)((tg * 64 + nth) * KT_G + kt)) * 512 + toff * 8;
    *reinterpret_cast<uint4*>(&sB[tid * 8]) =
        *reinterpret_cast<const uint4*>(src);
    __syncthreads();

    const __bf16* arow = (kt < 16) ? (x + (size_t)m * F_ + kt * 32)
                                   : (h + (size_t)m * H_ + (kt - 16) * 32);
    const v16bf a = load_a16(arow, sel);
    const v16bf b0 = load_b16(sb + 0 * 512 + lane * 16);
    const v16bf b1 = load_b16(sb + 1 * 512 + lane * 16);
    const v16bf b2 = load_b16(sb + 2 * 512 + lane * 16);
    const v16bf b3 = load_b16(sb + 3 * 512 + lane * 16);
    acc0 = wmma_bf16(a, b0, acc0);
    acc1 = wmma_bf16(a, b1, acc1);
    acc2 = wmma_bf16(a, b2, acc2);
    acc3 = wmma_bf16(a, b3, acc3);
    __syncthreads();
  }
#endif

  // epilogue: C layout -> lane holds column n = nth*16+lr; VGPR r -> row
  const int n = nth * 16 + lr;
  const float bi = bias[n];
  const float bf_ = bias[H_ + n];
  const float bg = bias[2 * H_ + n];
  const float bo = bias[3 * H_ + n];
#pragma unroll
  for (int r = 0; r < 8; ++r) {
    const int mm = wave * 16 + sel * 8 + r;
    const size_t idx = (size_t)mm * H_ + n;
    const float iv = sigmoidf_(acc0[r] + bi);
    const float fv = sigmoidf_(acc1[r] + bf_);
    const float gv = tanhf(acc2[r] + bg);
    const float ov = sigmoidf_(acc3[r] + bo);
    const float cn = fv * cbuf[idx] + iv * gv;
    cbuf[idx] = cn;
    hnew[idx] = to_bf16_bits(ov * tanhf(cn));
  }
}

// y = tanh(h @ fc_W^T + fc_b); write f32 to out[:,s,:] and bf16 to x feedback
// grid = 32 (F n-tiles), block = 256 (8 waves = 8 M-tiles)
__global__ __launch_bounds__(256) void k_fc(
    const unsigned short* __restrict__ hbits,   // [128,1024] bf16 (this step)
    const unsigned short* __restrict__ Wfbits,  // packed fc weights
    const float* __restrict__ fcb,              // [512]
    float* __restrict__ out,                    // [128,128,512] f32
    unsigned short* __restrict__ xnext,         // [128,512] bf16
    int s) {
  const int tid = threadIdx.x;
  const int wave = tid >> 5, lane = tid & 31;
  const int nt = blockIdx.x;                   // 0..31
  const int sel = lane >> 4, lr = lane & 15;
  const int m = wave * 16 + lr;

  const __bf16* h = reinterpret_cast<const __bf16*>(hbits);
  const __bf16* w = reinterpret_cast<const __bf16*>(Wfbits);

  v8f acc = {};

#if defined(USE_ASYNC_LDS)
  __shared__ __align__(16) unsigned short sB[3][512];       // 3 x 1KB
  const __bf16* sb = reinterpret_cast<const __bf16*>(&sB[0][0]);

  auto issue = [&](int kt, int slot) {
    // 256 threads x 4B = 1KB tile, uniform (no divergence)
    const __bf16* src = w + (size_t)(nt * KT_F + kt) * 512 + tid * 2;
    async_cp_b32(src, &sB[slot][tid * 2]);
  };

  issue(0, 0);
  int cur = 0;
  for (int kt = 0; kt < KT_F; ++kt) {
    int nxt = cur + 1; if (nxt == 3) nxt = 0;
    if (kt + 1 < KT_F) {
      issue(kt + 1, nxt);
      WAIT_ASYNC(1);
    } else {
      WAIT_ASYNC(0);
    }
    __syncthreads();
    const v16bf a = load_a16(h + (size_t)m * H_ + kt * 32, sel);
    const v16bf b = load_b16(sb + cur * 512 + lane * 16);
    acc = wmma_bf16(a, b, acc);
    cur = nxt;
  }
#else
  __shared__ __align__(16) unsigned short sB[512];
  const __bf16* sb = reinterpret_cast<const __bf16*>(sB);

  for (int kt = 0; kt < KT_F; ++kt) {
    reinterpret_cast<unsigned int*>(sB)[tid] = reinterpret_cast<const unsigned int*>(
        w + (size_t)(nt * KT_F + kt) * 512)[tid];
    __syncthreads();
    const v16bf a = load_a16(h + (size_t)m * H_ + kt * 32, sel);
    const v16bf b = load_b16(sb + lane * 16);
    acc = wmma_bf16(a, b, acc);
    __syncthreads();
  }
#endif

  const int n = nt * 16 + lr;
  const float bb = fcb[n];
#pragma unroll
  for (int r = 0; r < 8; ++r) {
    const int mm = wave * 16 + sel * 8 + r;
    const float y = tanhf(acc[r] + bb);
    out[((size_t)mm * S_ + s) * F_ + n] = y;     // out[b, s, f]
    xnext[(size_t)mm * F_ + n] = to_bf16_bits(y);
  }
}

// ---------------- launch ----------------

extern "C" void kernel_launch(void* const* d_in, const int* in_sizes, int n_in,
                              void* d_out, int out_size, void* d_ws,
                              size_t ws_size, hipStream_t stream) {
  const float* input = (const float*)d_in[0];
  const float* h0    = (const float*)d_in[1];
  const float* c0    = (const float*)d_in[2];
  const float* Wih   = (const float*)d_in[3];
  const float* Whh   = (const float*)d_in[4];
  const float* bih   = (const float*)d_in[5];
  const float* bhh   = (const float*)d_in[6];
  const float* fcW   = (const float*)d_in[7];
  const float* fcb   = (const float*)d_in[8];
  float* out = (float*)d_out;

  char* ws = (char*)d_ws;
  // workspace carve (256B aligned sizes)
  unsigned short* Wg  = (unsigned short*)(ws + 0);         // 4096*1536*2 = 12,582,912
  unsigned short* Wf  = (unsigned short*)(ws + 12582912);  //  512*1024*2 =  1,048,576
  float*          bias= (float*)        (ws + 13631488);   //      4096*4 =     16,384
  unsigned short* xb  = (unsigned short*)(ws + 13647872);  //   128*512*2 =    131,072
  unsigned short* hb0 = (unsigned short*)(ws + 13778944);  //  128*1024*2 =    262,144
  unsigned short* hb1 = (unsigned short*)(ws + 14041088);  //  128*1024*2 =    262,144
  float*          cb  = (float*)        (ws + 14303232);   //  128*1024*4 =    524,288
  // total ~14.8 MB

  k_pack_gates<<<(NG * KCAT) / 256, 256, 0, stream>>>(Wih, Whh, Wg);
  k_pack_fc<<<(F_ * H_) / 256, 256, 0, stream>>>(fcW, Wf);
  k_init<<<(B_ * H_) / 256, 256, 0, stream>>>(input, h0, c0, bih, bhh, bias, xb,
                                              hb0, cb);

  unsigned short* hbuf[2] = {hb0, hb1};
  for (int s = 0; s < S_; ++s) {
    unsigned short* hp = hbuf[s & 1];         // h from previous step
    unsigned short* hn = hbuf[(s + 1) & 1];   // h produced this step
    k_gates_cell<<<64, 256, 0, stream>>>(xb, hp, Wg, bias, cb, hn);
    k_fc<<<32, 256, 0, stream>>>(hn, Wf, fcb, out, xb, s);
  }
}
